// SFA_Gemma_15058155340462
// MI455X (gfx1250) — compile-verified, matrix operand order checked by reading, not run on hardware
//
#include <hip/hip_runtime.h>
#include <hip/hip_bf16.h>
#include <math.h>
#include <stdint.h>

// ---------------- model constants ----------------
constexpr int DMODEL = 3584;
constexpr int NH     = 16;
constexpr int KVH    = 8;
constexpr int DHD    = 256;
constexpr int FFN    = 8192;
constexpr int NOUT   = 320;
constexpr int NP1    = 4096;
constexpr int NP2    = 8192;
constexpr int NB     = 4;
constexpr int SEQ    = 512;
constexpr int TOK    = NB * SEQ;          // 2048
constexpr float LSCALE  = 4.0f;           // 16 / R
constexpr float QSCALE  = 0.0625f;        // 256^-0.5
constexpr float SOFTCAP = 50.0f;

// ---------------- bf16 helpers (bit-level) ----------------
__device__ __forceinline__ unsigned short f2bf(float f) {
  union { float f; unsigned u; } c; c.f = f;
  unsigned u = c.u;
  u += 0x7FFFu + ((u >> 16) & 1u);        // round to nearest even
  return (unsigned short)(u >> 16);
}
__device__ __forceinline__ float bf2f(unsigned short h) {
  union { unsigned u; float f; } c; c.u = ((unsigned)h) << 16;
  return c.f;
}

typedef __attribute__((ext_vector_type(16))) __bf16 v16bf;
typedef __attribute__((ext_vector_type(8)))  float  v8f;

union Frag16 { v16bf v; uint4 q[2]; unsigned short us[16]; };

// ---------------- CDNA5 async global->LDS copy (guarded) ----------------
#if defined(__has_builtin)
#  if __has_builtin(__builtin_amdgcn_global_load_async_to_lds_b128) && \
      __has_builtin(__builtin_amdgcn_s_wait_asynccnt)
#    define USE_ASYNC_LDS 1
#  endif
#endif
#ifndef USE_ASYNC_LDS
#  define USE_ASYNC_LDS 0
#endif

#if USE_ASYNC_LDS
typedef int v4i __attribute__((vector_size(16)));
typedef __attribute__((address_space(1))) v4i gv4i;   // global 16B vector
typedef __attribute__((address_space(3))) v4i lv4i;   // LDS 16B vector
#endif

__device__ __forceinline__ void cp16_to_lds(const unsigned short* g, unsigned short* l) {
#if USE_ASYNC_LDS
  __builtin_amdgcn_global_load_async_to_lds_b128(
      (gv4i*)(unsigned long long)(uintptr_t)g,
      (lv4i*)(unsigned)(uintptr_t)l, 0, 0);
#else
  *(uint4*)l = *(const uint4*)g;
#endif
}

__device__ __forceinline__ void wait_async_lds() {
#if USE_ASYNC_LDS
  __builtin_amdgcn_s_wait_asynccnt(0);
#endif
}

// =================================================================
// Batched WMMA GEMM:  C[m,n] = sum_k A[m,k] * B[n,k]  (+ rank-4 LoRA)
// A,B bf16, rows contiguous in K. Block = 8 wave32 -> 128x64 tile.
// K staged through double-buffered LDS in BK=64 chunks via async
// global->LDS b128 copies (ASYNCcnt) overlapping WMMA compute.
// LDS rows padded to 72 elements (144B) to spread DS b128 reads
// across all 64 banks.
// Batched over z = b*Hc + h; `rep` divides h for GQA on B.
// =================================================================
constexpr int BM  = 128;
constexpr int BN  = 64;
constexpr int BK  = 64;
constexpr int LDT = 72;   // padded LDS row stride in bf16 elements

template<bool USE_LORA>
__global__ __launch_bounds__(256) void k_gemm_bf16(
    const unsigned short* __restrict__ A, int lda, long sAb, long sAh,
    const unsigned short* __restrict__ B, int ldb, long sBb, long sBh, int rep,
    float* __restrict__ C, int ldc, long sCb, long sCh,
    int K, int Hc,
    const float* __restrict__ T4, const float* __restrict__ LB, float lscale)
{
  __shared__ __align__(16) unsigned short sA[2][BM * LDT];
  __shared__ __align__(16) unsigned short sB[2][BN * LDT];

  const int z  = blockIdx.z;
  const int bb = z / Hc;
  const int hh = z - bb * Hc;
  A += (size_t)bb * sAb + (size_t)hh * sAh;
  B += (size_t)bb * sBb + (size_t)(hh / rep) * sBh;
  C += (size_t)bb * sCb + (size_t)hh * sCh;

  const int tid  = threadIdx.x;
  const int lane = tid & 31;
  const int wave = tid >> 5;
  const int m0 = blockIdx.y * BM;
  const int n0 = blockIdx.x * BN;
  const unsigned short* Ag = A + (size_t)m0 * lda;
  const unsigned short* Bg = B + (size_t)n0 * ldb;

  // cooperative tile staging: 16B chunks, 8 chunks per 64-elem row
  auto stage = [&](int buf, int k0) {
    #pragma unroll
    for (int c = tid; c < BM * (BK / 8); c += 256) {
      const int row = c >> 3, sub = (c & 7) * 8;
      cp16_to_lds(Ag + (size_t)row * lda + k0 + sub, &sA[buf][row * LDT + sub]);
    }
    #pragma unroll
    for (int c = tid; c < BN * (BK / 8); c += 256) {
      const int row = c >> 3, sub = (c & 7) * 8;
      cp16_to_lds(Bg + (size_t)row * ldb + k0 + sub, &sB[buf][row * LDT + sub]);
    }
  };

  // fragment lane addressing per ISA 7.12.2 (16-bit A/B layouts)
  const int r16  = lane & 15;
  const int hi   = lane >> 4;
  const int kofs = hi * 8;

  v8f acc[4] = {};
  const int ntiles = K / BK;

  stage(0, 0);
  for (int t = 0; t < ntiles; ++t) {
    wait_async_lds();      // my async copies for buffer t&1 are done
    __syncthreads();       // everyone's copies are done / prev reads finished
    if (t + 1 < ntiles) stage((t + 1) & 1, (t + 1) * BK);

    const unsigned short* a_base = &sA[t & 1][0];
    const unsigned short* b_base = &sB[t & 1][0];
    #pragma unroll
    for (int kk = 0; kk < BK; kk += 32) {
      Frag16 fa, fb0, fb1, fb2, fb3;
      const unsigned short* ap = a_base + (wave * 16 + r16) * LDT + kk + kofs;
      fa.q[0]  = *(const uint4*)ap;
      fa.q[1]  = *(const uint4*)(ap + 16);
      const unsigned short* bp = b_base + r16 * LDT + kk + kofs;
      fb0.q[0] = *(const uint4*)(bp);
      fb0.q[1] = *(const uint4*)(bp + 16);
      fb1.q[0] = *(const uint4*)(bp + 16 * LDT);
      fb1.q[1] = *(const uint4*)(bp + 16 * LDT + 16);
      fb2.q[0] = *(const uint4*)(bp + 32 * LDT);
      fb2.q[1] = *(const uint4*)(bp + 32 * LDT + 16);
      fb3.q[0] = *(const uint4*)(bp + 48 * LDT);
      fb3.q[1] = *(const uint4*)(bp + 48 * LDT + 16);
      acc[0] = __builtin_amdgcn_wmma_f32_16x16x32_bf16(false, fa.v, false, fb0.v, (short)0, acc[0], false, false);
      acc[1] = __builtin_amdgcn_wmma_f32_16x16x32_bf16(false, fa.v, false, fb1.v, (short)0, acc[1], false, false);
      acc[2] = __builtin_amdgcn_wmma_f32_16x16x32_bf16(false, fa.v, false, fb2.v, (short)0, acc[2], false, false);
      acc[3] = __builtin_amdgcn_wmma_f32_16x16x32_bf16(false, fa.v, false, fb3.v, (short)0, acc[3], false, false);
    }
  }

  // C/D layout: lane&15 = N, VGPR r = M = r + 8*(lane>>4)
  #pragma unroll
  for (int j = 0; j < 4; ++j) {
    const int n = n0 + j * 16 + r16;
    #pragma unroll
    for (int r = 0; r < 8; ++r) {
      const int m = m0 + wave * 16 + hi * 8 + r;
      float val = acc[j][r];
      if constexpr (USE_LORA) {
        const float* tm = T4 + (size_t)m * 4;
        const float* lb = LB + (size_t)n * 4;
        val += lscale * (tm[0]*lb[0] + tm[1]*lb[1] + tm[2]*lb[2] + tm[3]*lb[3]);
      }
      C[(size_t)m * ldc + n] = val;
    }
  }
}

// ---------------- embedding gather + scale ----------------
__global__ void k_embed(const int* __restrict__ X, const float* __restrict__ E,
                        float* __restrict__ h)
{
  const int t = blockIdx.x;
  const float sc = sqrtf((float)DMODEL);
  const float* er = E + (size_t)X[t] * DMODEL;
  float* hr = h + (size_t)t * DMODEL;
  for (int d = threadIdx.x; d < DMODEL; d += 256) hr[d] = er[d] * sc;
}

// ---------------- RMSNorm (MODE 0: ->bf16, 1: ->f32, 2: add into f32) ----------------
template<int MODE>
__global__ void k_rms(const float* __restrict__ y, const float* __restrict__ g,
                      void* __restrict__ out, int Dn)
{
  __shared__ float red[256];
  const size_t base = (size_t)blockIdx.x * Dn;
  float ss = 0.f;
  for (int d = threadIdx.x; d < Dn; d += 256) { float v = y[base + d]; ss += v * v; }
  red[threadIdx.x] = ss; __syncthreads();
  for (int s = 128; s > 0; s >>= 1) {
    if (threadIdx.x < s) red[threadIdx.x] += red[threadIdx.x + s];
    __syncthreads();
  }
  const float rstd = rsqrtf(red[0] / (float)Dn + 1e-6f);
  for (int d = threadIdx.x; d < Dn; d += 256) {
    float v = y[base + d] * rstd * (1.0f + g[d]);
    if constexpr (MODE == 0)      ((unsigned short*)out)[base + d] = f2bf(v);
    else if constexpr (MODE == 1) ((float*)out)[base + d] = v;
    else                          ((float*)out)[base + d] += v;
  }
}

// ---------------- LoRA: t[m,0..3] = sum_k x_bf16[m,k] * A[r,k] ----------------
__global__ void k_lora_t(const unsigned short* __restrict__ x, const float* __restrict__ A,
                         float* __restrict__ t, int K)
{
  __shared__ float red[256];
  const size_t base = (size_t)blockIdx.x * K;
  float a0 = 0, a1 = 0, a2 = 0, a3 = 0;
  for (int k = threadIdx.x; k < K; k += 256) {
    float xv = bf2f(x[base + k]);
    a0 += xv * A[k];
    a1 += xv * A[(size_t)K + k];
    a2 += xv * A[(size_t)2 * K + k];
    a3 += xv * A[(size_t)3 * K + k];
  }
  float acc[4] = {a0, a1, a2, a3};
  for (int r = 0; r < 4; ++r) {
    red[threadIdx.x] = acc[r]; __syncthreads();
    for (int s = 128; s > 0; s >>= 1) {
      if (threadIdx.x < s) red[threadIdx.x] += red[threadIdx.x + s];
      __syncthreads();
    }
    if (threadIdx.x == 0) t[(size_t)blockIdx.x * 4 + r] = red[0];
    __syncthreads();
  }
}

// ---------------- RoPE (f32 in -> bf16 out), layout [t, head, dh] ----------------
__global__ void k_rope(const float* __restrict__ src, unsigned short* __restrict__ dst, int Hn)
{
  const int id = blockIdx.x * blockDim.x + threadIdx.x; // TOK*Hn*128 threads
  const int i  = id & 127;
  const int hd = (id >> 7) % Hn;
  const int t  = id / (128 * Hn);
  const int s  = t & (SEQ - 1);
  const size_t base = ((size_t)t * Hn + hd) * DHD;
  const float inv = __expf(-(float)i * (9.210340372f / 128.0f)); // 10000^(-i/128)
  float sn, cs;
  __sincosf((float)s * inv, &sn, &cs);
  const float t1 = src[base + i], t2 = src[base + 128 + i];
  dst[base + i]       = f2bf(t1 * cs - t2 * sn);
  dst[base + 128 + i] = f2bf(t2 * cs + t1 * sn);
}

// ---------------- v [b,s,kv,dh] f32 -> vT [b,kv,dh,s] bf16 ----------------
__global__ void k_vT(const float* __restrict__ vf, unsigned short* __restrict__ vT)
{
  const int id = blockIdx.x * blockDim.x + threadIdx.x; // TOK*KVH*DHD
  const int dh = id & 255;
  const int kv = (id >> 8) & 7;
  const int t  = id >> 11;
  const int b  = t >> 9;
  const int s  = t & 511;
  vT[(((size_t)(b * KVH + kv) * DHD) + dh) * SEQ + s] = f2bf(vf[id]);
}

// ---------------- softcap + causal softmax, one wave per score row ----------------
__global__ __launch_bounds__(256) void k_softmax_cap(const float* __restrict__ sc,
                                                     unsigned short* __restrict__ p)
{
  const int row  = blockIdx.x * 8 + (threadIdx.x >> 5); // [b,h,q] flattened
  const int lane = threadIdx.x & 31;
  const int qpos = row & (SEQ - 1);
  const float* r = sc + (size_t)row * SEQ;
  float v[16];
  float mx = -1e30f;
  #pragma unroll
  for (int j = 0; j < 16; ++j) {
    const int e = j * 32 + lane;
    float s = r[e] * QSCALE;
    s = SOFTCAP * tanhf(s * (1.0f / SOFTCAP));
    if (e > qpos) s = -1e30f;
    v[j] = s;
    mx = fmaxf(mx, s);
  }
  for (int o = 16; o > 0; o >>= 1) mx = fmaxf(mx, __shfl_xor(mx, o, 32));
  float sum = 0.f;
  #pragma unroll
  for (int j = 0; j < 16; ++j) { v[j] = __expf(v[j] - mx); sum += v[j]; }
  for (int o = 16; o > 0; o >>= 1) sum += __shfl_xor(sum, o, 32);
  const float inv = 1.0f / sum;
  unsigned short* pr = p + (size_t)row * SEQ;
  #pragma unroll
  for (int j = 0; j < 16; ++j) pr[j * 32 + lane] = f2bf(v[j] * inv);
}

// ---------------- elementwise f32 -> bf16 ----------------
__global__ void k_cvt_bf16(const float* __restrict__ s, unsigned short* __restrict__ d, size_t n)
{
  const size_t i = (size_t)blockIdx.x * blockDim.x + threadIdx.x;
  if (i < n) d[i] = f2bf(s[i]);
}

// ---------------- gelu(tanh)(g) * u -> bf16 ----------------
__global__ void k_gelu_mul(const float* __restrict__ g, const float* __restrict__ u,
                           unsigned short* __restrict__ d, size_t n)
{
  const size_t i = (size_t)blockIdx.x * blockDim.x + threadIdx.x;
  if (i >= n) return;
  const float x = g[i];
  const float ge = 0.5f * x * (1.0f + tanhf(0.7978845608f * (x + 0.044715f * x * x * x)));
  d[i] = f2bf(ge * u[i]);
}

// ---------------- mean pool over SEQ ----------------
__global__ void k_pool(const float* __restrict__ hf, float* __restrict__ pooled)
{
  const int id = blockIdx.x * blockDim.x + threadIdx.x; // NB*DMODEL
  const int b = id / DMODEL, d = id % DMODEL;
  float s = 0.f;
  for (int q = 0; q < SEQ; ++q) s += hf[((size_t)(b * SEQ + q)) * DMODEL + d];
  pooled[id] = s * (1.0f / (float)SEQ);
}

// ---------------- one wave per output: out[b,n] = x[b,:]·W[n,:] + bias ----------------
__global__ void k_dot(const float* __restrict__ x, const float* __restrict__ W,
                      const float* __restrict__ bias, float* __restrict__ out,
                      float* __restrict__ out2, int Brows, int N, int K)
{
  const int widx = blockIdx.x * 8 + (threadIdx.x >> 5);
  if (widx >= Brows * N) return;
  const int lane = threadIdx.x & 31;
  const int b = widx / N, n = widx % N;
  const float* xr = x + (size_t)b * K;
  const float* wr = W + (size_t)n * K;
  float acc = 0.f;
  for (int k = lane; k < K; k += 32) acc += xr[k] * wr[k];
  for (int o = 16; o > 0; o >>= 1) acc += __shfl_xor(acc, o, 32);
  if (lane == 0) {
    const float r = acc + (bias ? bias[n] : 0.0f);
    out[widx] = r;
    if (out2) out2[widx] = r;
  }
}

// ---------------- batchnorm over batch (4 samples) + relu, in place ----------------
__global__ void k_bn_relu(float* __restrict__ h1, const float* __restrict__ g,
                          const float* __restrict__ b)
{
  const int n = blockIdx.x * blockDim.x + threadIdx.x;
  if (n >= NP1) return;
  const float x0 = h1[n], x1 = h1[NP1 + n], x2 = h1[2 * NP1 + n], x3 = h1[3 * NP1 + n];
  const float mu  = 0.25f * (x0 + x1 + x2 + x3);
  const float var = 0.25f * (x0 * x0 + x1 * x1 + x2 * x2 + x3 * x3) - mu * mu;
  const float rs  = rsqrtf(var + 1e-5f);
  const float gg = g[n], bb = b[n];
  h1[n]           = fmaxf(0.f, (x0 - mu) * rs * gg + bb);
  h1[NP1 + n]     = fmaxf(0.f, (x1 - mu) * rs * gg + bb);
  h1[2 * NP1 + n] = fmaxf(0.f, (x2 - mu) * rs * gg + bb);
  h1[3 * NP1 + n] = fmaxf(0.f, (x3 - mu) * rs * gg + bb);
}

// =================================================================
// host side
// =================================================================
static inline void launch_gemm(hipStream_t st,
    const void* A, int lda, long sAb, long sAh,
    const void* B, int ldb, long sBb, long sBh, int rep,
    float* C, int ldc, long sCb, long sCh,
    int M, int N, int K, int Hc, int nz,
    const float* T4, const float* LB, float ls)
{
  dim3 grid(N / BN, M / BM, nz);
  dim3 blk(256);
  if (T4)
    k_gemm_bf16<true><<<grid, blk, 0, st>>>((const unsigned short*)A, lda, sAb, sAh,
        (const unsigned short*)B, ldb, sBb, sBh, rep, C, ldc, sCb, sCh, K, Hc, T4, LB, ls);
  else
    k_gemm_bf16<false><<<grid, blk, 0, st>>>((const unsigned short*)A, lda, sAb, sAh,
        (const unsigned short*)B, ldb, sBb, sBh, rep, C, ldc, sCb, sCh, K, Hc, nullptr, nullptr, 0.f);
}

extern "C" void kernel_launch(void* const* d_in, const int* in_sizes, int n_in,
                              void* d_out, int out_size, void* d_ws, size_t ws_size,
                              hipStream_t stream)
{
  (void)in_sizes; (void)n_in; (void)out_size; (void)ws_size;

  // ---- inputs (setup_inputs dict order) ----
  const int*   X       = (const int*)  d_in[0];
  const float* embed   = (const float*)d_in[1];
  const float* ln_in   = (const float*)d_in[2];
  const float* ln_pa   = (const float*)d_in[3];
  const float* ln_pf   = (const float*)d_in[4];
  const float* ln_pff  = (const float*)d_in[5];
  const float* ln_fin  = (const float*)d_in[6];
  const float* Wq = (const float*)d_in[7],  *Aq = (const float*)d_in[8],  *Bq = (const float*)d_in[9];
  const float* Wk = (const float*)d_in[10], *Ak = (const float*)d_in[11], *Bk = (const float*)d_in[12];
  const float* Wv = (const float*)d_in[13], *Av = (const float*)d_in[14], *Bv = (const float*)d_in[15];
  const float* Wo = (const float*)d_in[16], *Ao = (const float*)d_in[17], *Bo = (const float*)d_in[18];
  const float* Wg = (const float*)d_in[19], *Ag = (const float*)d_in[20], *Bg = (const float*)d_in[21];
  const float* Wu = (const float*)d_in[22], *Au = (const float*)d_in[23], *Bu = (const float*)d_in[24];
  const float* Wd = (const float*)d_in[25], *Ad = (const float*)d_in[26], *Bd = (const float*)d_in[27];
  const float* lin_w = (const float*)d_in[28], *lin_b = (const float*)d_in[29];
  const float* pw1 = (const float*)d_in[30], *pb1 = (const float*)d_in[31];
  const float* bn_g = (const float*)d_in[32], *bn_b = (const float*)d_in[33];
  const float* pw2 = (const float*)d_in[34];
  float* out = (float*)d_out;

  // ---- workspace layout ----
  char* w = (char*)d_ws;
  size_t off = 0;
  auto alloc = [&](size_t bytes) -> void* {
    off = (off + 255) & ~(size_t)255;
    void* p = w + off;
    off += bytes;
    return p;
  };
  float*          h_f    = (float*)alloc((size_t)TOK * DMODEL * 4);
  unsigned short* xb     = (unsigned short*)alloc((size_t)TOK * DMODEL * 2);
  float*          qf     = (float*)alloc((size_t)TOK * NH * DHD * 4);      // reused as `of`
  float*          kf     = (float*)alloc((size_t)TOK * KVH * DHD * 4);     // reused as `ob` (bf16)
  float*          vf     = (float*)alloc((size_t)TOK * KVH * DHD * 4);
  unsigned short* qb     = (unsigned short*)alloc((size_t)TOK * NH * DHD * 2);
  unsigned short* kbuf   = (unsigned short*)alloc((size_t)TOK * KVH * DHD * 2);
  unsigned short* vT     = (unsigned short*)alloc((size_t)TOK * KVH * DHD * 2);
  float*          scores = (float*)alloc((size_t)NB * NH * SEQ * SEQ * 4); // reused as ubuf
  unsigned short* pbuf   = (unsigned short*)alloc((size_t)NB * NH * SEQ * SEQ * 2); // reused as prod
  float*          gbuf   = (float*)alloc((size_t)TOK * FFN * 4);
  float*          ybuf   = (float*)alloc((size_t)TOK * DMODEL * 4);        // attn_out / dbuf / hf
  unsigned short* wqb = (unsigned short*)alloc((size_t)NH * DHD * DMODEL * 2);
  unsigned short* wkb = (unsigned short*)alloc((size_t)KVH * DHD * DMODEL * 2);
  unsigned short* wvb = (unsigned short*)alloc((size_t)KVH * DHD * DMODEL * 2);
  unsigned short* wob = (unsigned short*)alloc((size_t)DMODEL * NH * DHD * 2);
  unsigned short* wgb = (unsigned short*)alloc((size_t)FFN * DMODEL * 2);
  unsigned short* wub = (unsigned short*)alloc((size_t)FFN * DMODEL * 2);
  unsigned short* wdb = (unsigned short*)alloc((size_t)DMODEL * FFN * 2);
  float* tq = (float*)alloc(TOK * 4 * 4);
  float* tk = (float*)alloc(TOK * 4 * 4);
  float* tv = (float*)alloc(TOK * 4 * 4);
  float* to = (float*)alloc(TOK * 4 * 4);
  float* tg = (float*)alloc(TOK * 4 * 4);
  float* tu = (float*)alloc(TOK * 4 * 4);
  float* td = (float*)alloc(TOK * 4 * 4);
  float* pooled = (float*)alloc(NB * DMODEL * 4);
  float* xo     = (float*)alloc(NB * NOUT * 4);
  float* h1     = (float*)alloc(NB * NP1 * 4);

  float*          of_  = qf;                    // pv output reuses qf
  unsigned short* ob   = (unsigned short*)kf;   // bf16 attn output reuses kf
  float*          ubuf = scores;                // up-proj reuses score buffer
  unsigned short* prod = pbuf;                  // gelu(g)*u reuses p buffer

  auto cvt = [&](const float* s, unsigned short* d, size_t n) {
    k_cvt_bf16<<<dim3((unsigned)((n + 255) / 256)), dim3(256), 0, stream>>>(s, d, n);
  };

  // ---- weights f32 -> bf16 once per call (amortized over 16 M-tile reuses) ----
  cvt(Wq, wqb, (size_t)NH * DHD * DMODEL);
  cvt(Wk, wkb, (size_t)KVH * DHD * DMODEL);
  cvt(Wv, wvb, (size_t)KVH * DHD * DMODEL);
  cvt(Wo, wob, (size_t)DMODEL * NH * DHD);
  cvt(Wg, wgb, (size_t)FFN * DMODEL);
  cvt(Wu, wub, (size_t)FFN * DMODEL);
  cvt(Wd, wdb, (size_t)DMODEL * FFN);

  // ---- embed + first RMSNorm ----
  k_embed<<<dim3(TOK), dim3(256), 0, stream>>>(X, embed, h_f);
  k_rms<0><<<dim3(TOK), dim3(256), 0, stream>>>(h_f, ln_in, xb, DMODEL);

  // ---- LoRA low-rank activations for q/k/v ----
  k_lora_t<<<dim3(TOK), dim3(256), 0, stream>>>(xb, Aq, tq, DMODEL);
  k_lora_t<<<dim3(TOK), dim3(256), 0, stream>>>(xb, Ak, tk, DMODEL);
  k_lora_t<<<dim3(TOK), dim3(256), 0, stream>>>(xb, Av, tv, DMODEL);

  // ---- q/k/v projections (WMMA) ----
  launch_gemm(stream, xb, DMODEL, 0, 0, wqb, DMODEL, 0, 0, 1,
              qf, NH * DHD, 0, 0, TOK, NH * DHD, DMODEL, 1, 1, tq, Bq, LSCALE);
  launch_gemm(stream, xb, DMODEL, 0, 0, wkb, DMODEL, 0, 0, 1,
              kf, KVH * DHD, 0, 0, TOK, KVH * DHD, DMODEL, 1, 1, tk, Bk, LSCALE);
  launch_gemm(stream, xb, DMODEL, 0, 0, wvb, DMODEL, 0, 0, 1,
              vf, KVH * DHD, 0, 0, TOK, KVH * DHD, DMODEL, 1, 1, tv, Bv, LSCALE);

  // ---- RoPE + v transpose (all -> bf16) ----
  k_rope<<<dim3(TOK * NH * 128 / 256), dim3(256), 0, stream>>>(qf, qb, NH);
  k_rope<<<dim3(TOK * KVH * 128 / 256), dim3(256), 0, stream>>>(kf, kbuf, KVH);
  k_vT<<<dim3(TOK * KVH * DHD / 256), dim3(256), 0, stream>>>(vf, vT);

  // ---- attention scores: q·k^T, batched over (b,h), GQA rep=2 ----
  launch_gemm(stream,
              qb, NH * DHD, (long)SEQ * NH * DHD, (long)DHD,
              kbuf, KVH * DHD, (long)SEQ * KVH * DHD, (long)DHD, 2,
              scores, SEQ, (long)NH * SEQ * SEQ, (long)SEQ * SEQ,
              SEQ, SEQ, DHD, NH, NB * NH, nullptr, nullptr, 0.f);

  // ---- softcap + causal softmax -> p (bf16) ----
  k_softmax_cap<<<dim3(NB * NH * SEQ / 8), dim3(256), 0, stream>>>(scores, pbuf);

  // ---- p·v^T -> o [b,s,h,dh] ----
  launch_gemm(stream,
              pbuf, SEQ, (long)NH * SEQ * SEQ, (long)SEQ * SEQ,
              vT, SEQ, (long)KVH * DHD * SEQ, (long)DHD * SEQ, 2,
              of_, NH * DHD, (long)SEQ * NH * DHD, (long)DHD,
              SEQ, DHD, SEQ, NH, NB * NH, nullptr, nullptr, 0.f);

  // ---- output projection + post-attn RMSNorm residual ----
  cvt(of_, ob, (size_t)TOK * NH * DHD);
  k_lora_t<<<dim3(TOK), dim3(256), 0, stream>>>(ob, Ao, to, NH * DHD);
  launch_gemm(stream, ob, NH * DHD, 0, 0, wob, NH * DHD, 0, 0, 1,
              ybuf, DMODEL, 0, 0, TOK, DMODEL, NH * DHD, 1, 1, to, Bo, LSCALE);
  k_rms<2><<<dim3(TOK), dim3(256), 0, stream>>>(ybuf, ln_pa, h_f, DMODEL);

  // ---- FFN ----
  k_rms<0><<<dim3(TOK), dim3(256), 0, stream>>>(h_f, ln_pf, xb, DMODEL);
  k_lora_t<<<dim3(TOK), dim3(256), 0, stream>>>(xb, Ag, tg, DMODEL);
  k_lora_t<<<dim3(TOK), dim3(256), 0, stream>>>(xb, Au, tu, DMODEL);
  launch_gemm(stream, xb, DMODEL, 0, 0, wgb, DMODEL, 0, 0, 1,
              gbuf, FFN, 0, 0, TOK, FFN, DMODEL, 1, 1, tg, Bg, LSCALE);
  launch_gemm(stream, xb, DMODEL, 0, 0, wub, DMODEL, 0, 0, 1,
              ubuf, FFN, 0, 0, TOK, FFN, DMODEL, 1, 1, tu, Bu, LSCALE);
  k_gelu_mul<<<dim3((unsigned)((size_t)TOK * FFN / 256)), dim3(256), 0, stream>>>(
      gbuf, ubuf, prod, (size_t)TOK * FFN);
  k_lora_t<<<dim3(TOK), dim3(256), 0, stream>>>(prod, Ad, td, FFN);
  launch_gemm(stream, prod, FFN, 0, 0, wdb, FFN, 0, 0, 1,
              ybuf, DMODEL, 0, 0, TOK, DMODEL, FFN, 1, 1, td, Bd, LSCALE);
  k_rms<2><<<dim3(TOK), dim3(256), 0, stream>>>(ybuf, ln_pff, h_f, DMODEL);

  // ---- final RMSNorm + mean pool ----
  k_rms<1><<<dim3(TOK), dim3(256), 0, stream>>>(h_f, ln_fin, ybuf, DMODEL);
  k_pool<<<dim3(NB * DMODEL / 256), dim3(256), 0, stream>>>(ybuf, pooled);

  // ---- head (tiny M -> bandwidth-bound f32 GEMVs) ----
  k_dot<<<dim3(NB * NOUT / 8), dim3(256), 0, stream>>>(pooled, lin_w, lin_b, xo, out,
                                                       NB, NOUT, DMODEL);
  k_dot<<<dim3(NB * NP1 / 8), dim3(256), 0, stream>>>(xo, pw1, pb1, h1, nullptr,
                                                      NB, NP1, NOUT);
  k_bn_relu<<<dim3(NP1 / 256), dim3(256), 0, stream>>>(h1, bn_g, bn_b);
  k_dot<<<dim3(NB * NP2 / 8), dim3(256), 0, stream>>>(h1, pw2, nullptr, out + NB * NOUT,
                                                      nullptr, NB, NP2, NP1);
}
